// Abode_uncond_64630667870367
// MI455X (gfx1250) — matrix operand order repaired and checked by reading, not compiled, not run on hardware
//
#include <hip/hip_runtime.h>
#include <math.h>

// ---------------------------------------------------------------------------
// Types for CDNA5 WMMA
// ---------------------------------------------------------------------------
typedef __attribute__((ext_vector_type(16))) __bf16   v16bf;
typedef __attribute__((ext_vector_type(8)))  __bf16   bf8v;
typedef __attribute__((ext_vector_type(16))) _Float16 v16h;
typedef __attribute__((ext_vector_type(8)))  _Float16 h8v;
typedef __attribute__((ext_vector_type(8)))  float    v8f;

#define NN 50000
#define NE 800000

__device__ __forceinline__ v8f wmma_bf16(v16bf a, v16bf b, v8f c) {
  return __builtin_amdgcn_wmma_f32_16x16x32_bf16(false, a, false, b, (short)0, c,
                                                 false, false);
}
__device__ __forceinline__ v8f wmma_f16(v16h a, v16h b, v8f c) {
  return __builtin_amdgcn_wmma_f32_16x16x32_f16(false, a, false, b, (short)0, c,
                                                false, false);
}

// A-matrix (16x32, 16-bit) loader from row-major [rows][ldx]. ISA layout:
// lanes 0-15 hold M=lane, elems 0..7 = K k0+0..7, elems 8..15 = K k0+16..23;
// lanes 16-31 hold M=lane-16, elems 0..7 = K k0+8..15, 8..15 = K k0+24..31.
template <typename V16, typename V8, typename T>
__device__ __forceinline__ V16 loadA(const T* __restrict__ X, int ldx, int m0,
                                     int k0, int lane) {
  int row  = m0 + (lane & 15);
  int koff = (lane & 16) ? 8 : 0;
  const T* p = X + (size_t)row * ldx + k0 + koff;
  V8 lo = *(const V8*)(p);
  V8 hi = *(const V8*)(p + 16);
  V16 a;
#pragma unroll
  for (int i = 0; i < 8; ++i) { a[i] = lo[i]; a[i + 8] = hi[i]; }
  return a;
}

// B fragment load from pre-swizzled weights: layout [nt][kt][lane][16] so each
// lane's 16 elements are contiguous (single 32B vector load, no addr chains).
template <typename V16, typename T>
__device__ __forceinline__ V16 loadBswz(const T* __restrict__ WD, int Kt,
                                        int nt, int kt, int lane) {
  const T* p = WD + ((((size_t)nt * Kt + kt) * 32) + lane) * 16;
  return *(const V16*)p;
}

// Order-preserving float<->uint for atomicMax-based segment max.
__device__ __forceinline__ unsigned f2ord(float f) {
  unsigned b = __float_as_uint(f);
  return (b & 0x80000000u) ? ~b : (b | 0x80000000u);
}
__device__ __forceinline__ float ord2f(unsigned u) {
  unsigned b = (u & 0x80000000u) ? (u & 0x7FFFFFFFu) : ~u;
  return __uint_as_float(b);
}

__device__ __forceinline__ void nrm3(float& x, float& y, float& z) {
  float n   = sqrtf(x * x + y * y + z * z);
  float inv = 1.0f / fmaxf(n, 1e-12f);
  x *= inv; y *= inv; z *= inv;
}

// ---------------------------------------------------------------------------
// Per-node orientation frames (rows: o1, n2, o1 x n2); last node = 0.
// ---------------------------------------------------------------------------
__global__ void k_orient(const float* __restrict__ data, float* __restrict__ O,
                         int n) {
  int i = blockIdx.x * blockDim.x + threadIdx.x;
  if (i >= n) return;
  float* o = O + (size_t)i * 9;
  if (i == n - 1) {
#pragma unroll
    for (int j = 0; j < 9; ++j) o[j] = 0.0f;
    return;
  }
  const float* c = data + (size_t)i * 29 + 20;
  float ax = c[3] - c[0], ay = c[4] - c[1], az = c[5] - c[2];  // u2
  float bx = c[6] - c[3], by = c[7] - c[4], bz = c[8] - c[5];  // u1
  nrm3(ax, ay, az);
  nrm3(bx, by, bz);
  float nx = ay * bz - az * by, ny = az * bx - ax * bz, nz = ax * by - ay * bx;
  nrm3(nx, ny, nz);
  float ox = ax - bx, oy = ay - by, oz = az - bz;
  nrm3(ox, oy, oz);
  float tx = oy * nz - oz * ny, ty = oz * nx - ox * nz, tz = ox * ny - oy * nx;
  o[0] = ox; o[1] = oy; o[2] = oz;
  o[3] = nx; o[4] = ny; o[5] = nz;
  o[6] = tx; o[7] = ty; o[8] = tz;
}

// ---------------------------------------------------------------------------
// Edge features (45 dims) -> f16, K-padded to 64 for WMMA.
// ---------------------------------------------------------------------------
__global__ void k_edge_feat(const float* __restrict__ data,
                            const int* __restrict__ eidx,
                            const int* __restrict__ amino,
                            const float* __restrict__ O,
                            _Float16* __restrict__ attr, int ne) {
  int e = blockIdx.x * blockDim.x + threadIdx.x;
  if (e >= ne) return;
  int s = eidx[e];
  int d = eidx[ne + e];
  const float* xs = data + (size_t)s * 29;
  const float* xd = data + (size_t)d * 29;
  float f[45];
  f[0] = (float)amino[s] - (float)amino[d];
#pragma unroll
  for (int j = 0; j < 20; ++j) f[1 + j] = xs[j] - xd[j];
  float diff[9], rv[9];
#pragma unroll
  for (int j = 0; j < 9; ++j) diff[j] = xs[20 + j] - xd[20 + j];
#pragma unroll
  for (int row = 0; row < 3; ++row) {
    float dx = diff[row * 3], dy = diff[row * 3 + 1], dz = diff[row * 3 + 2];
    float r  = sqrtf(dx * dx + dy * dy + dz * dz);
    f[21 + row] = __expf(-r);
    float inv = 1.0f / fmaxf(r, 1e-12f);
    rv[row * 3] = dx * inv; rv[row * 3 + 1] = dy * inv; rv[row * 3 + 2] = dz * inv;
  }
#pragma unroll
  for (int j = 0; j < 9; ++j) f[24 + j] = rv[j];
  const float* Os = O + (size_t)s * 9;
  const float* Od = O + (size_t)d * 9;
#pragma unroll
  for (int i = 0; i < 3; ++i)
#pragma unroll
    for (int k = 0; k < 3; ++k)
      f[33 + i * 3 + k] =
          Os[i] * Od[k] + Os[3 + i] * Od[3 + k] + Os[6 + i] * Od[6 + k];
  float vx = rv[3], vy = rv[4], vz = rv[5];
  nrm3(vx, vy, vz);
#pragma unroll
  for (int i = 0; i < 3; ++i)
    f[42 + i] = Os[i] * vx + Os[3 + i] * vy + Os[6 + i] * vz;
  _Float16* out = attr + (size_t)e * 64;
#pragma unroll
  for (int j = 0; j < 45; ++j) out[j] = (_Float16)f[j];
#pragma unroll
  for (int j = 45; j < 64; ++j) out[j] = (_Float16)0.0f;
}

// ---------------------------------------------------------------------------
// Packing kernels
// ---------------------------------------------------------------------------
__global__ void k_pack_x0(const float* __restrict__ data,
                          const int* __restrict__ tptr,
                          __bf16* __restrict__ XP, int n) {
  int i = blockIdx.x * blockDim.x + threadIdx.x;
  if (i >= n) return;
  float t = (float)tptr[0];
  __bf16* row = XP + (size_t)i * 32;
  row[0] = (__bf16)t;
#pragma unroll
  for (int j = 0; j < 29; ++j) row[1 + j] = (__bf16)data[(size_t)i * 29 + j];
  row[30] = (__bf16)0.0f;
  row[31] = (__bf16)0.0f;
}

// Pack fp32 weights [cin][cout] into the WMMA B-fragment-swizzled layout
// [Npad/16][Kpad/32][32 lanes][16], zero-padded in both K and N.
__global__ void k_pack_w_swz_bf(const float* __restrict__ W,
                                __bf16* __restrict__ WD, int cin, int cout,
                                int Kpad, int Npad) {
  long idx   = (long)blockIdx.x * blockDim.x + threadIdx.x;
  long total = (long)(Npad >> 4) * (Kpad >> 5) * 512;
  if (idx >= total) return;
  int  i    = (int)(idx & 15);
  int  lane = (int)((idx >> 4) & 31);
  long t    = idx >> 9;
  int  Kt   = Kpad >> 5;
  int  kt   = (int)(t % Kt);
  int  nt   = (int)(t / Kt);
  int  col  = nt * 16 + (lane & 15);
  int  k    = kt * 32 + ((lane & 16) ? 8 : 0) + (i & 7) + ((i & 8) ? 16 : 0);
  float v   = (col < cout && k < cin) ? W[(size_t)k * cout + col] : 0.0f;
  WD[idx] = (__bf16)v;
}

__global__ void k_pack_w_swz_h(const float* __restrict__ W,
                               _Float16* __restrict__ WD, int cin, int cout,
                               int Kpad, int Npad) {
  long idx   = (long)blockIdx.x * blockDim.x + threadIdx.x;
  long total = (long)(Npad >> 4) * (Kpad >> 5) * 512;
  if (idx >= total) return;
  int  i    = (int)(idx & 15);
  int  lane = (int)((idx >> 4) & 31);
  long t    = idx >> 9;
  int  Kt   = Kpad >> 5;
  int  kt   = (int)(t % Kt);
  int  nt   = (int)(t / Kt);
  int  col  = nt * 16 + (lane & 15);
  int  k    = kt * 32 + ((lane & 16) ? 8 : 0) + (i & 7) + ((i & 8) ? 16 : 0);
  float v   = (col < cout && k < cin) ? W[(size_t)k * cout + col] : 0.0f;
  WD[idx] = (_Float16)v;
}

// act: 0 = relu, 1 = sigmoid
__global__ void k_pack_act(const float* __restrict__ H, __bf16* __restrict__ XP,
                           int n, int cin, int Kpad, int act) {
  long idx = (long)blockIdx.x * blockDim.x + threadIdx.x;
  if (idx >= (long)n * Kpad) return;
  int i = (int)(idx / Kpad), c = (int)(idx % Kpad);
  float v = 0.0f;
  if (c < cin) {
    v = H[(size_t)i * cin + c];
    v = (act == 0) ? fmaxf(v, 0.0f) : 1.0f / (1.0f + __expf(-v));
  }
  XP[idx] = (__bf16)v;
}

__global__ void k_init_ms(unsigned* __restrict__ mbuf, float* __restrict__ sbuf,
                          int n) {
  int i = blockIdx.x * blockDim.x + threadIdx.x;
  if (i >= n) return;
  mbuf[i] = 0u;
  sbuf[i] = 0.0f;
}

// ---------------------------------------------------------------------------
// Node GEMM: one wave = 16x64 output tile (4 WMMA accumulators) of one of
// {q,k,v,out=x@Ws}. A fragment reused 4x; B fragments are single vector loads.
// ---------------------------------------------------------------------------
__global__ void k_node_gemm(const __bf16* __restrict__ XP,
                            const __bf16* __restrict__ WP,
                            const float* __restrict__ bq,
                            const float* __restrict__ bk,
                            const float* __restrict__ bv,
                            const float* __restrict__ bs,
                            float* __restrict__ qb, float* __restrict__ kb,
                            float* __restrict__ vb, float* __restrict__ ob,
                            int n, int Kpad, int cout, int ntM, int ntG) {
  int wave = (int)((blockIdx.x * blockDim.x + threadIdx.x) >> 5);
  int lane = threadIdx.x & 31;
  int tpm  = ntM * ntG;
  int mat  = wave / tpm;
  if (mat >= 4) return;  // wave-uniform
  int rem = wave - mat * tpm;
  int m0  = (rem / ntG) * 16;
  int g   = rem % ntG;  // group of 4 column tiles (64 cols)
  int Kt  = Kpad >> 5;
  int NtT = ntG * 4;  // total padded col tiles
  const __bf16* W = WP + (size_t)mat * Kt * NtT * 512;
  v8f c0 = {0.f, 0.f, 0.f, 0.f, 0.f, 0.f, 0.f, 0.f};
  v8f c1 = c0, c2 = c0, c3 = c0;
  for (int kt = 0; kt < Kt; ++kt) {
    v16bf a = loadA<v16bf, bf8v>(XP, Kpad, m0, kt * 32, lane);
    c0 = wmma_bf16(a, loadBswz<v16bf>(W, Kt, g * 4 + 0, kt, lane), c0);
    c1 = wmma_bf16(a, loadBswz<v16bf>(W, Kt, g * 4 + 1, kt, lane), c1);
    c2 = wmma_bf16(a, loadBswz<v16bf>(W, Kt, g * 4 + 2, kt, lane), c2);
    c3 = wmma_bf16(a, loadBswz<v16bf>(W, Kt, g * 4 + 3, kt, lane), c3);
  }
  const float* bias = (mat == 0) ? bq : (mat == 1) ? bk : (mat == 2) ? bv : bs;
  float* dst        = (mat == 0) ? qb : (mat == 1) ? kb : (mat == 2) ? vb : ob;
  int rbase = m0 + ((lane & 16) ? 8 : 0);
  v8f acc[4] = {c0, c1, c2, c3};
#pragma unroll
  for (int t = 0; t < 4; ++t) {
    int col = (g * 4 + t) * 16 + (lane & 15);
    if (col < cout) {
      float bvl = bias[col];
#pragma unroll
      for (int r = 0; r < 8; ++r) {
        int row = rbase + r;
        if (row < n) dst[(size_t)row * cout + col] = acc[t][r] + bvl;
      }
    }
  }
}

// ---------------------------------------------------------------------------
// Pass 1: alpha = dot(q[dst], k[src] + attr@We)/sqrt(c); segment max (atomic).
// One wave = 16 edges; e recomputed via f16 WMMA (never materialized).
// ---------------------------------------------------------------------------
__global__ void k_edge_alpha(const _Float16* __restrict__ attr,
                             const _Float16* __restrict__ WeP,
                             const float* __restrict__ qb,
                             const float* __restrict__ kb,
                             const int* __restrict__ eidx,
                             float* __restrict__ alpha,
                             unsigned* __restrict__ mbuf, int ne, int cout,
                             float scale) {
  int wave = (int)((blockIdx.x * blockDim.x + threadIdx.x) >> 5);
  int lane = threadIdx.x & 31;
  int e0   = wave * 16;
  if (e0 >= ne) return;  // wave-uniform
  int mbase = (lane & 16) ? 8 : 0;
  int srcI[8], dstI[8];
#pragma unroll
  for (int r = 0; r < 8; ++r) {
    int e   = e0 + mbase + r;
    srcI[r] = eidx[e];
    dstI[r] = eidx[ne + e];
  }
  v16h a0 = loadA<v16h, h8v>(attr, 64, e0, 0, lane);
  v16h a1 = loadA<v16h, h8v>(attr, 64, e0, 32, lane);
  float acc[8] = {0, 0, 0, 0, 0, 0, 0, 0};
  int col0 = lane & 15;
  int ntN  = (cout + 15) >> 4;
  for (int nt = 0; nt < ntN; ++nt) {
    v8f c = {0.f, 0.f, 0.f, 0.f, 0.f, 0.f, 0.f, 0.f};
    c = wmma_f16(a0, loadBswz<v16h>(WeP, 2, nt, 0, lane), c);
    c = wmma_f16(a1, loadBswz<v16h>(WeP, 2, nt, 1, lane), c);
    int col = nt * 16 + col0;
    if (col < cout) {
#pragma unroll
      for (int r = 0; r < 8; ++r) {
        float qd = qb[(size_t)dstI[r] * cout + col];
        float ks = kb[(size_t)srcI[r] * cout + col];
        acc[r] += qd * (ks + c[r]);
      }
    }
  }
#pragma unroll
  for (int r = 0; r < 8; ++r) {
    float x = acc[r];
    x += __shfl_xor(x, 1, 32);
    x += __shfl_xor(x, 2, 32);
    x += __shfl_xor(x, 4, 32);
    x += __shfl_xor(x, 8, 32);
    acc[r] = x * scale;
  }
  if ((lane & 15) == 0) {
#pragma unroll
    for (int r = 0; r < 8; ++r) {
      int e    = e0 + mbase + r;
      alpha[e] = acc[r];
      atomicMax(&mbuf[dstI[r]], f2ord(acc[r]));
    }
  }
}

// Pass 2: ex = exp(alpha - m[dst]); segment sum (atomic).
__global__ void k_edge_ex(const float* __restrict__ alpha,
                          const unsigned* __restrict__ mbuf,
                          const int* __restrict__ eidx, float* __restrict__ exb,
                          float* __restrict__ sbuf, int ne) {
  int e = blockIdx.x * blockDim.x + threadIdx.x;
  if (e >= ne) return;
  int d    = eidx[ne + e];
  float ex = __expf(alpha[e] - ord2f(mbuf[d]));
  exb[e]   = ex;
  atomicAdd(&sbuf[d], ex);
}

// Pass 3: out[dst] += (v[src] + attr@We) * ex/(s[dst]+eps). e recomputed.
__global__ void k_edge_out(const _Float16* __restrict__ attr,
                           const _Float16* __restrict__ WeP,
                           const float* __restrict__ vb,
                           const float* __restrict__ exb,
                           const float* __restrict__ sbuf,
                           const int* __restrict__ eidx,
                           float* __restrict__ outp, int ne, int cout) {
  int wave = (int)((blockIdx.x * blockDim.x + threadIdx.x) >> 5);
  int lane = threadIdx.x & 31;
  int e0   = wave * 16;
  if (e0 >= ne) return;  // wave-uniform
  int mbase = (lane & 16) ? 8 : 0;
  int srcI[8], dstI[8];
  float coef[8];
#pragma unroll
  for (int r = 0; r < 8; ++r) {
    int e   = e0 + mbase + r;
    srcI[r] = eidx[e];
    dstI[r] = eidx[ne + e];
    coef[r] = exb[e] / (sbuf[dstI[r]] + 1e-16f);
  }
  v16h a0 = loadA<v16h, h8v>(attr, 64, e0, 0, lane);
  v16h a1 = loadA<v16h, h8v>(attr, 64, e0, 32, lane);
  int col0 = lane & 15;
  int ntN  = (cout + 15) >> 4;
  for (int nt = 0; nt < ntN; ++nt) {
    v8f c = {0.f, 0.f, 0.f, 0.f, 0.f, 0.f, 0.f, 0.f};
    c = wmma_f16(a0, loadBswz<v16h>(WeP, 2, nt, 0, lane), c);
    c = wmma_f16(a1, loadBswz<v16h>(WeP, 2, nt, 1, lane), c);
    int col = nt * 16 + col0;
    if (col < cout) {
#pragma unroll
      for (int r = 0; r < 8; ++r) {
        float vs = vb[(size_t)srcI[r] * cout + col];
        atomicAdd(&outp[(size_t)dstI[r] * cout + col], (vs + c[r]) * coef[r]);
      }
    }
  }
}

// ---------------------------------------------------------------------------
// Host orchestration
// ---------------------------------------------------------------------------
extern "C" void kernel_launch(void* const* d_in, const int* in_sizes, int n_in,
                              void* d_out, int out_size, void* d_ws,
                              size_t ws_size, hipStream_t stream) {
  (void)in_sizes; (void)n_in; (void)out_size; (void)ws_size;
  const float* data  = (const float*)d_in[0];
  const int*   tptr  = (const int*)d_in[1];
  const int*   eidx  = (const int*)d_in[2];
  const int*   amino = (const int*)d_in[3];

  static const int CIN[4]  = {30, 128, 256, 64};
  static const int COUT[4] = {128, 256, 64, 29};
  static const int KPAD[4] = {32, 128, 256, 64};
  static const int ACT[4]  = {0, 1, 1, -1};

  char*  ws  = (char*)d_ws;
  size_t off = 0;
  auto alloc = [&](size_t b) -> void* {
    void* p = ws + off;
    off = (off + b + 255) & ~(size_t)255;
    return p;
  };
  __bf16*   XP    = (__bf16*)alloc((size_t)NN * 256 * 2);
  __bf16*   WP    = (__bf16*)alloc((size_t)4 * 256 * 256 * 2);
  _Float16* WeP   = (_Float16*)alloc((size_t)64 * 256 * 2);
  float*    qb    = (float*)alloc((size_t)NN * 256 * 4);
  float*    kb    = (float*)alloc((size_t)NN * 256 * 4);
  float*    vb    = (float*)alloc((size_t)NN * 256 * 4);
  float*    ob    = (float*)alloc((size_t)NN * 256 * 4);
  float*    Obuf  = (float*)alloc((size_t)NN * 9 * 4);
  _Float16* attr  = (_Float16*)alloc((size_t)NE * 64 * 2);
  float*    alpha = (float*)alloc((size_t)NE * 4);
  float*    exb   = (float*)alloc((size_t)NE * 4);
  unsigned* mbuf  = (unsigned*)alloc((size_t)NN * 4);
  float*    sbuf  = (float*)alloc((size_t)NN * 4);

  const int BT = 256;
  auto cdiv = [](long a, long b) { return (int)((a + b - 1) / b); };

  k_orient<<<cdiv(NN, BT), BT, 0, stream>>>(data, Obuf, NN);
  k_edge_feat<<<cdiv(NE, BT), BT, 0, stream>>>(data, eidx, amino, Obuf, attr, NE);
  k_pack_x0<<<cdiv(NN, BT), BT, 0, stream>>>(data, tptr, XP, NN);

  for (int l = 0; l < 4; ++l) {
    int cin = CIN[l], cout = COUT[l], Kpad = KPAD[l];
    int ntG   = (cout + 63) / 64;   // groups of 4 col tiles per wave
    int Npad  = ntG * 64;           // node-weight padded width
    int NpadE = ((cout + 15) / 16) * 16;  // We padded width
    const float* Wm[4];
    Wm[0]           = (const float*)d_in[4 + l * 9 + 0];  // Wq
    Wm[1]           = (const float*)d_in[4 + l * 9 + 1];  // Wk
    Wm[2]           = (const float*)d_in[4 + l * 9 + 2];  // Wv
    Wm[3]           = (const float*)d_in[4 + l * 9 + 3];  // Ws
    const float* We = (const float*)d_in[4 + l * 9 + 4];
    const float* bq = (const float*)d_in[4 + l * 9 + 5];
    const float* bk = (const float*)d_in[4 + l * 9 + 6];
    const float* bv = (const float*)d_in[4 + l * 9 + 7];
    const float* bs = (const float*)d_in[4 + l * 9 + 8];

    long wtot = (long)(Npad >> 4) * (Kpad >> 5) * 512;
    for (int m = 0; m < 4; ++m)
      k_pack_w_swz_bf<<<cdiv(wtot, BT), BT, 0, stream>>>(
          Wm[m], WP + (size_t)m * wtot, cin, cout, Kpad, Npad);
    long wetot = (long)(NpadE >> 4) * 2 * 512;
    k_pack_w_swz_h<<<cdiv(wetot, BT), BT, 0, stream>>>(We, WeP, 45, cout, 64,
                                                       NpadE);

    float* outp = (l == 3) ? (float*)d_out : ob;
    int ntM = NN / 16;
    long waves = 4L * ntM * ntG;
    k_node_gemm<<<cdiv(waves * 32, BT), BT, 0, stream>>>(
        XP, WP, bq, bk, bv, bs, qb, kb, vb, outp, NN, Kpad, cout, ntM, ntG);

    k_init_ms<<<cdiv(NN, BT), BT, 0, stream>>>(mbuf, sbuf, NN);
    long ewaves = NE / 16;
    float scale = 1.0f / sqrtf((float)cout);
    k_edge_alpha<<<cdiv(ewaves * 32, BT), BT, 0, stream>>>(
        attr, WeP, qb, kb, eidx, alpha, mbuf, NE, cout, scale);
    k_edge_ex<<<cdiv(NE, BT), BT, 0, stream>>>(alpha, mbuf, eidx, exb, sbuf, NE);
    k_edge_out<<<cdiv(ewaves * 32, BT), BT, 0, stream>>>(
        attr, WeP, vb, exb, sbuf, eidx, outp, NE, cout);

    if (l < 3)
      k_pack_act<<<cdiv((long)NN * KPAD[l + 1], BT), BT, 0, stream>>>(
          outp, XP, NN, cout, KPAD[l + 1], ACT[l]);
  }
}
